// CCM_73985106641118
// MI455X (gfx1250) — matrix operand order repaired and checked by reading, not compiled, not run on hardware
//
#include <hip/hip_runtime.h>
#include <hip/hip_bf16.h>
#include <math.h>

#define DEVI __device__ __forceinline__

typedef __attribute__((ext_vector_type(16))) __bf16          v16bf;
typedef __attribute__((ext_vector_type(8)))  float           v8f;
typedef __attribute__((ext_vector_type(8)))  unsigned short  u16x8;
typedef __attribute__((ext_vector_type(4)))  unsigned int    u32x4;
typedef __attribute__((ext_vector_type(4)))  int             i32x4;
typedef __attribute__((ext_vector_type(8)))  int             i32x8;

union Frag16 {
    unsigned short s[16];
    u16x8          h8[2];
    v16bf          v;
};

DEVI unsigned short f2bf(float f) {
    unsigned int u = __float_as_uint(f);
    u += 0x7FFFu + ((u >> 16) & 1u);
    return (unsigned short)(u >> 16);
}

// Problem constants (B,N,C,H,K from the reference)
static constexpr int Bb  = 8;
static constexpr int Nt  = 2048;
static constexpr int Cc  = 512;
static constexpr int Hh  = 512;
static constexpr int Kc  = 64;
static constexpr int BN  = Bb * Nt;   // 16384 token rows

// ---------------------------------------------------------------------------
// Tensor Data Mover: load a 2-D tile (rows x width bf16, row stride = width)
// from global into LDS.  D# per CDNA5 ISA ch.8.
// ---------------------------------------------------------------------------
DEVI void tdm_load_panel(unsigned ldsOff, const unsigned short* gsrc,
                         int width /*elems*/, int rows)
{
    unsigned long long ga = (unsigned long long)(const void*)gsrc;
    u32x4 g0;
    g0[0] = 1u;                                              // count=1, no gather
    g0[1] = ldsOff;                                          // lds_addr (bytes)
    g0[2] = (unsigned)ga;                                    // global_addr[31:0]
    g0[3] = (unsigned)((ga >> 32) & 0x01FFFFFFu) | (2u << 30); // addr[56:32]|type=2
    i32x8 g1;
    g1[0] = (int)(1u << 16);                                 // data_size=1 -> 2B
    g1[1] = (int)(((unsigned)width & 0xFFFFu) << 16);        // tensor_dim0[15:0]
    g1[2] = (int)((((unsigned)width >> 16) & 0xFFFFu)        // tensor_dim0[31:16]
                  | (((unsigned)rows & 0xFFFFu) << 16));     // tensor_dim1[15:0]
    g1[3] = (int)((((unsigned)rows >> 16) & 0xFFFFu)         // tensor_dim1[31:16]
                  | (((unsigned)width & 0xFFFFu) << 16));    // tile_dim0
    g1[4] = (int)((unsigned)rows & 0xFFFFu);                 // tile_dim1 (tile_dim2=0)
    g1[5] = width;                                           // tensor_dim0_stride lo32
    g1[6] = 0;
    g1[7] = 0;
    i32x4 gz = {0, 0, 0, 0};
#if __clang_major__ >= 23
    i32x8 gz8 = {0, 0, 0, 0, 0, 0, 0, 0};
    __builtin_amdgcn_tensor_load_to_lds(g0, g1, gz, gz, gz8, 0);
#else
    __builtin_amdgcn_tensor_load_to_lds(g0, g1, gz, gz, 0);
#endif
}

DEVI v8f wmma_bf16(const Frag16& a, const Frag16& b, v8f c) {
    return __builtin_amdgcn_wmma_f32_16x16x32_bf16(
        false, a.v, false, b.v, (short)0, c, false, false);
}

// ---------------------------------------------------------------------------
// LDS-staged WMMA GEMM (Bt contiguous layout only).
// Block = 8 waves; each wave owns a 32(M) x 64(N) tile -> block = 256 M-rows.
// The 64 x Kd bf16 B panel (<=64KB) is DMA'd into LDS once by the TDM; all
// waves feed B fragments with ds_load_b128 (each shared by 2 WMMAs).  A
// fragments are ping-pong double-buffered from global (K unroll x2, no
// register copies -> no WMMA hazard nops).
// Requires M % 256 == 0, Kd % 64 == 0, Kd <= 512.
// ---------------------------------------------------------------------------
template<bool RELU>
__global__ void __launch_bounds__(256)
k_gemm_lds(const unsigned short* __restrict__ A0, int lda, long sA,
           const unsigned short* __restrict__ Bt0, long sB,
           const float* __restrict__ bias, const float* __restrict__ scaleVec,
           float scaleK,
           float* __restrict__ oF0, unsigned short* __restrict__ oB0, int ldo,
           long sO, int M, int Ncols, int Kd)
{
    __shared__ __attribute__((aligned(16))) unsigned short Bpanel[64 * 512];
    const int lane = threadIdx.x & 31;
    const int wv   = threadIdx.x >> 5;
    const int nGroups = Ncols >> 6;
    const int mBlk = blockIdx.x / nGroups;
    const int nGrp = blockIdx.x - mBlk * nGroups;
    const int m0 = (mBlk * 8 + wv) * 32;       // 32 rows per wave
    const int n0 = nGrp * 64;
    const int z  = blockIdx.y;
    const unsigned short* A  = A0  + (size_t)z * sA;
    const unsigned short* Bt = Bt0 + (size_t)z * sB;

    if (wv == 0) {
        tdm_load_panel((unsigned)(size_t)Bpanel, Bt + (size_t)n0 * Kd, Kd, 64);
        __builtin_amdgcn_s_wait_tensorcnt(0);
    }
    __syncthreads();

    v8f acc[2][4];
    #pragma unroll
    for (int mt = 0; mt < 2; ++mt)
        #pragma unroll
        for (int t = 0; t < 4; ++t)
            #pragma unroll
            for (int v = 0; v < 8; ++v) acc[mt][t][v] = 0.0f;

    const int kbA = (lane >> 4) << 3;
    const int nIx = lane & 15;
    const int kbB = (lane >> 4) << 3;
    const unsigned short* pa0 = A + (size_t)(m0 + (lane & 15)) * lda + kbA;
    const unsigned short* pa1 = pa0 + (size_t)16 * lda;

    Frag16 aA[2], aB[2];
    aA[0].h8[0] = *(const u16x8*)(pa0);
    aA[0].h8[1] = *(const u16x8*)(pa0 + 16);
    aA[1].h8[0] = *(const u16x8*)(pa1);
    aA[1].h8[1] = *(const u16x8*)(pa1 + 16);

    for (int k0 = 0; k0 < Kd; k0 += 64) {
        // prefetch A frags for k0+32 (always in range: Kd % 64 == 0)
        aB[0].h8[0] = *(const u16x8*)(pa0 + k0 + 32);
        aB[0].h8[1] = *(const u16x8*)(pa0 + k0 + 48);
        aB[1].h8[0] = *(const u16x8*)(pa1 + k0 + 32);
        aB[1].h8[1] = *(const u16x8*)(pa1 + k0 + 48);
        #pragma unroll
        for (int t = 0; t < 4; ++t) {
            Frag16 fb;
            const unsigned short* pb = &Bpanel[(size_t)(t * 16 + nIx) * Kd + k0 + kbB];
            fb.h8[0] = *(const u16x8*)(pb);
            fb.h8[1] = *(const u16x8*)(pb + 16);
            acc[0][t] = wmma_bf16(aA[0], fb, acc[0][t]);
            acc[1][t] = wmma_bf16(aA[1], fb, acc[1][t]);
        }
        if (k0 + 64 < Kd) {     // prefetch A frags for k0+64 into the A buffer
            aA[0].h8[0] = *(const u16x8*)(pa0 + k0 + 64);
            aA[0].h8[1] = *(const u16x8*)(pa0 + k0 + 80);
            aA[1].h8[0] = *(const u16x8*)(pa1 + k0 + 64);
            aA[1].h8[1] = *(const u16x8*)(pa1 + k0 + 80);
        }
        #pragma unroll
        for (int t = 0; t < 4; ++t) {
            Frag16 fb;
            const unsigned short* pb = &Bpanel[(size_t)(t * 16 + nIx) * Kd + k0 + 32 + kbB];
            fb.h8[0] = *(const u16x8*)(pb);
            fb.h8[1] = *(const u16x8*)(pb + 16);
            acc[0][t] = wmma_bf16(aB[0], fb, acc[0][t]);
            acc[1][t] = wmma_bf16(aB[1], fb, acc[1][t]);
        }
    }

    float*          oF = oF0 ? oF0 + (size_t)z * sO : nullptr;
    unsigned short* oB = oB0 ? oB0 + (size_t)z * sO : nullptr;
    const int mHalf = (lane >> 4) << 3;
    #pragma unroll
    for (int mt = 0; mt < 2; ++mt) {
        #pragma unroll
        for (int t = 0; t < 4; ++t) {
            const int nc = n0 + t * 16 + nIx;
            const float sv = scaleK * (scaleVec ? scaleVec[nc] : 1.0f);
            const float bv = bias ? bias[nc] : 0.0f;
            #pragma unroll
            for (int v = 0; v < 8; ++v) {
                const int m = m0 + mt * 16 + mHalf + v;
                float val = acc[mt][t][v] * sv + bv;
                if (RELU) val = fmaxf(val, 0.0f);
                if (oF) oF[(size_t)m * ldo + nc] = val;
                if (oB) oB[(size_t)m * ldo + nc] = f2bf(val);
            }
        }
    }
}

// ---------------------------------------------------------------------------
// Generic WMMA GEMM (global-fed).  Used for the small / row-major-B cases.
// ---------------------------------------------------------------------------
template<bool BT, bool RELU>
__global__ void __launch_bounds__(256)
k_gemm(const unsigned short* __restrict__ A0, int lda, long sA,
       const unsigned short* __restrict__ B0, int ldb, long sB,
       const float* __restrict__ bias, const float* __restrict__ scaleVec,
       float scaleK,
       float* __restrict__ oF0, unsigned short* __restrict__ oB0, int ldo, long sO,
       unsigned short* __restrict__ oT0, int ldt, long sT,
       int M, int Ncols, int Kd)
{
    const int lane    = threadIdx.x & 31;
    const int wave    = blockIdx.x * (blockDim.x >> 5) + (threadIdx.x >> 5);
    const int nGroups = Ncols >> 6;
    const int mTile   = wave / nGroups;
    if (mTile * 16 >= M) return;           // whole-wave uniform: EXEC stays all-1s
    const int nGrp = wave - mTile * nGroups;
    const int m0 = mTile * 16, n0 = nGrp * 64;

    const int z = blockIdx.y;
    const unsigned short* A  = A0 + (size_t)z * sA;
    const unsigned short* Bm = B0 + (size_t)z * sB;

    v8f acc[4];
    #pragma unroll
    for (int t = 0; t < 4; ++t)
        #pragma unroll
        for (int v = 0; v < 8; ++v) acc[t][v] = 0.0f;

    const int mA  = m0 + (lane & 15);
    const int kbA = (lane >> 4) << 3;
    const int nIx = lane & 15;
    const int kbB = (lane >> 4) << 3;

    for (int k0 = 0; k0 < Kd; k0 += 32) {
        Frag16 fa;
        const unsigned short* pa = A + (size_t)mA * lda + k0 + kbA;
        fa.h8[0] = *(const u16x8*)(pa);
        fa.h8[1] = *(const u16x8*)(pa + 16);
        if (k0 + 32 < Kd) __builtin_prefetch(pa + 32, 0, 1);
        #pragma unroll
        for (int t = 0; t < 4; ++t) {
            Frag16 fb;
            const int nc = n0 + t * 16 + nIx;
            if (BT) {
                const unsigned short* pb = Bm + (size_t)nc * ldb + k0 + kbB;
                fb.h8[0] = *(const u16x8*)(pb);
                fb.h8[1] = *(const u16x8*)(pb + 16);
            } else {
                #pragma unroll
                for (int e = 0; e < 16; ++e)
                    fb.s[e] = Bm[(size_t)(k0 + kbB + (e & 7) + ((e >> 3) << 4)) * ldb + nc];
            }
            acc[t] = wmma_bf16(fa, fb, acc[t]);
        }
    }

    float*          oF = oF0 ? oF0 + (size_t)z * sO : nullptr;
    unsigned short* oB = oB0 ? oB0 + (size_t)z * sO : nullptr;
    unsigned short* oT = oT0 ? oT0 + (size_t)z * sT : nullptr;
    const int mHalf = (lane >> 4) << 3;
    #pragma unroll
    for (int t = 0; t < 4; ++t) {
        const int nc = n0 + t * 16 + nIx;
        const float sv = scaleK * (scaleVec ? scaleVec[nc] : 1.0f);
        const float bv = bias ? bias[nc] : 0.0f;
        #pragma unroll
        for (int v = 0; v < 8; ++v) {
            const int m = m0 + mHalf + v;
            float val = acc[t][v] * sv + bv;
            if (RELU) val = fmaxf(val, 0.0f);
            if (oF) oF[(size_t)m * ldo + nc] = val;
            if (oB) oB[(size_t)m * ldo + nc] = f2bf(val);
            if (oT) oT[(size_t)nc * ldt + m] = f2bf(val);
        }
    }
}

// ---------------------------------------------------------------------------
// x row L2-normalize (over C=512) + convert to bf16.  One wave per row.
// ---------------------------------------------------------------------------
__global__ void __launch_bounds__(256)
k_normalize(const float* __restrict__ x, unsigned short* __restrict__ xn)
{
    const int row  = blockIdx.x * 8 + (threadIdx.x >> 5);
    const int lane = threadIdx.x & 31;
    const float* xr = x + (size_t)row * Cc;
    float vals[16], s = 0.0f;
    #pragma unroll
    for (int j = 0; j < 16; ++j) { float v = xr[lane + 32 * j]; vals[j] = v; s += v * v; }
    #pragma unroll
    for (int m = 16; m; m >>= 1) s += __shfl_xor(s, m, 32);
    const float inv = 1.0f / fmaxf(sqrtf(s), 1e-12f);
    unsigned short* o = xn + (size_t)row * Cc;
    #pragma unroll
    for (int j = 0; j < 16; ++j) o[lane + 32 * j] = f2bf(vals[j] * inv);
}

// f32 [R][Cn] -> bf16 [Cn][R]   (weight -> Bt layout)
__global__ void __launch_bounds__(256)
k_transpose_cvt(const float* __restrict__ in, unsigned short* __restrict__ out,
                int R, int Cn)
{
    const int i = blockIdx.x * 256 + threadIdx.x;
    if (i >= R * Cn) return;
    const int r = i / Cn, c = i - r * Cn;
    out[(size_t)c * R + r] = f2bf(in[i]);
}

// plain f32 -> bf16
__global__ void __launch_bounds__(256)
k_cvt_bf(const float* __restrict__ in, unsigned short* __restrict__ out, int n)
{
    const int i = blockIdx.x * 256 + threadIdx.x;
    if (i < n) out[i] = f2bf(in[i]);
}

// per-cluster 1/||E_k||  (64 rows x 512)
__global__ void k_enorm(const float* __restrict__ E, float* __restrict__ invEn)
{
    const int row = blockIdx.x, lane = threadIdx.x;
    const float* r = E + (size_t)row * Hh;
    float s = 0.0f;
    #pragma unroll
    for (int j = 0; j < 16; ++j) { float v = r[lane + 32 * j]; s += v * v; }
    #pragma unroll
    for (int m = 16; m; m >>= 1) s += __shfl_xor(s, m, 32);
    if (lane == 0) invEn[row] = 1.0f / sqrtf(s);
}

// softmax over K=64 per token + bernoulli mask.  One wave per token.
__global__ void __launch_bounds__(256)
k_softmax64(const float* __restrict__ Lg, const float* __restrict__ bern,
            float* __restrict__ P, unsigned short* __restrict__ Pbf,
            unsigned short* __restrict__ Mbf)
{
    const int tok  = blockIdx.x * 8 + (threadIdx.x >> 5);
    const int lane = threadIdx.x & 31;
    const size_t base = (size_t)tok * Kc;
    const float v0 = Lg[base + lane], v1 = Lg[base + lane + 32];
    float mx = fmaxf(v0, v1);
    #pragma unroll
    for (int m = 16; m; m >>= 1) mx = fmaxf(mx, __shfl_xor(mx, m, 32));
    const float e0 = __expf(v0 - mx), e1 = __expf(v1 - mx);
    float s = e0 + e1;
    #pragma unroll
    for (int m = 16; m; m >>= 1) s += __shfl_xor(s, m, 32);
    const float inv = 1.0f / s;
    const float p0 = e0 * inv, p1 = e1 * inv;
    P[base + lane] = p0;           P[base + lane + 32] = p1;
    Pbf[base + lane] = f2bf(p0);   Pbf[base + lane + 32] = f2bf(p1);
    Mbf[base + lane]      = (bern[base + lane]      < p0) ? (unsigned short)0x3F80u : 0u;
    Mbf[base + lane + 32] = (bern[base + lane + 32] < p1) ? (unsigned short)0x3F80u : 0u;
}

// softmax over row length `n` -> bf16.  One 256-thread block per row.
__global__ void __launch_bounds__(256)
k_softmax_row(const float* __restrict__ S, unsigned short* __restrict__ Abf, int n)
{
    const int row = blockIdx.x;
    const float* s = S + (size_t)row * n;
    unsigned short* o = Abf + (size_t)row * n;
    __shared__ float redm[8], reds[8];
    const int tid = threadIdx.x, lane = tid & 31, wv = tid >> 5;
    float mx = -3.4e38f;
    for (int i = tid; i < n; i += 256) mx = fmaxf(mx, s[i]);
    #pragma unroll
    for (int m = 16; m; m >>= 1) mx = fmaxf(mx, __shfl_xor(mx, m, 32));
    if (lane == 0) redm[wv] = mx;
    __syncthreads();
    mx = redm[0];
    #pragma unroll
    for (int i = 1; i < 8; ++i) mx = fmaxf(mx, redm[i]);
    float sum = 0.0f;
    for (int i = tid; i < n; i += 256) sum += __expf(s[i] - mx);
    #pragma unroll
    for (int m = 16; m; m >>= 1) sum += __shfl_xor(sum, m, 32);
    if (lane == 0) reds[wv] = sum;
    __syncthreads();
    sum = 0.0f;
    #pragma unroll
    for (int i = 0; i < 8; ++i) sum += reds[i];
    const float inv = 1.0f / sum;
    for (int i = tid; i < n; i += 256) o[i] = f2bf(__expf(s[i] - mx) * inv);
}

// L2-normalize C_pre[b][h][:] over n, store transposed into outC[b][n][h].
__global__ void __launch_bounds__(256)
k_colnorm(const float* __restrict__ Cpre, float* __restrict__ outC, int n)
{
    const int row = blockIdx.x;              // b*H + h
    const int b = row >> 9, h = row & 511;
    const float* src = Cpre + (size_t)row * n;
    __shared__ float red[8];
    const int tid = threadIdx.x, lane = tid & 31, wv = tid >> 5;
    float ss = 0.0f;
    for (int i = tid; i < n; i += 256) { float v = src[i]; ss += v * v; }
    #pragma unroll
    for (int m = 16; m; m >>= 1) ss += __shfl_xor(ss, m, 32);
    if (lane == 0) red[wv] = ss;
    __syncthreads();
    ss = 0.0f;
    #pragma unroll
    for (int i = 0; i < 8; ++i) ss += red[i];
    const float inv = 1.0f / fmaxf(sqrtf(ss), 1e-12f);
    float* dst = outC + ((size_t)b * Nt) * Hh + h;
    for (int i = tid; i < n; i += 256) dst[(size_t)i * Hh] = src[i] * inv;
}

// U = bf16( (Hm + C) * theta )
__global__ void __launch_bounds__(256)
k_build_U(const float* __restrict__ Hm, const float* __restrict__ Cn,
          const float* __restrict__ theta, unsigned short* __restrict__ U)
{
    const size_t i = (size_t)blockIdx.x * 256 + threadIdx.x;
    U[i] = f2bf((Hm[i] + Cn[i]) * theta[i]);
}

// ---------------------------------------------------------------------------
extern "C" void kernel_launch(void* const* d_in, const int* in_sizes, int n_in,
                              void* d_out, int out_size, void* d_ws, size_t ws_size,
                              hipStream_t stream)
{
    (void)in_sizes; (void)n_in; (void)out_size; (void)ws_size;
    const float* x      = (const float*)d_in[0];
    const float* bern   = (const float*)d_in[1];
    const float* E      = (const float*)d_in[2];
    const float* w1     = (const float*)d_in[3];
    const float* b1     = (const float*)d_in[4];
    const float* w2     = (const float*)d_in[5];
    const float* b2     = (const float*)d_in[6];
    const float* wq     = (const float*)d_in[7];
    const float* wq_b   = (const float*)d_in[8];
    const float* wk     = (const float*)d_in[9];
    const float* wk_b   = (const float*)d_in[10];
    const float* wout   = (const float*)d_in[11];
    const float* wout_b = (const float*)d_in[12];

    float* Y    = (float*)d_out;                       // (B,N,C) first
    float* outC = (float*)d_out + (size_t)BN * Cc;     // (B,N,H) second

    char* ws = (char*)d_ws;
    size_t off = 0;
    auto take = [&](size_t bytes) -> void* {
        void* p = ws + off;
        off = (off + bytes + 255) & ~(size_t)255;
        return p;
    };
    unsigned short* xn     = (unsigned short*)take((size_t)BN * Cc * 2);
    unsigned short* w1t    = (unsigned short*)take((size_t)Cc * Hh * 2);
    unsigned short* w2t    = (unsigned short*)take((size_t)Hh * Hh * 2);
    unsigned short* wkt    = (unsigned short*)take((size_t)Hh * Hh * 2);
    unsigned short* wqt    = (unsigned short*)take((size_t)Hh * Hh * 2);
    unsigned short* woutt  = (unsigned short*)take((size_t)Hh * Cc * 2);
    unsigned short* Ebf    = (unsigned short*)take((size_t)Kc * Hh * 2);
    float*          invEn  = (float*)         take((size_t)Kc * 4);
    unsigned short* T1     = (unsigned short*)take((size_t)BN * Hh * 2);
    float*          Hm     = (float*)         take((size_t)BN * Hh * 4);
    unsigned short* Hmbf   = (unsigned short*)take((size_t)BN * Hh * 2);
    float*          Lg     = (float*)         take((size_t)BN * Kc * 4);
    float*          P      = (float*)         take((size_t)BN * Kc * 4);
    unsigned short* Pbf    = (unsigned short*)take((size_t)BN * Kc * 2);
    unsigned short* Mbf    = (unsigned short*)take((size_t)BN * Kc * 2);
    float*          theta  = (float*)         take((size_t)BN * Hh * 4);
    unsigned short* Qbf    = (unsigned short*)take((size_t)Kc * Hh * 2);
    float*          scores = (float*)         take((size_t)Bb * Kc * Nt * 4);
    unsigned short* Abf    = (unsigned short*)take((size_t)Bb * Kc * Nt * 2);
    unsigned short* CtT    = (unsigned short*)take((size_t)Bb * Hh * Kc * 2);
    float*          Cpre   = (float*)         take((size_t)Bb * Hh * Nt * 4);
    // Safe aliases (lifetimes disjoint): xn dead after GEMM1, T1 dead after GEMM2
    unsigned short* Kmatbf = xn;
    unsigned short* Ubf    = T1;

    const dim3 blk(256);
    auto gb  = [](int M, int Nc) { return dim3(((M / 16) * (Nc / 64) + 7) / 8); };
    auto gbl = [](int M, int Nc) { return dim3((M / 256) * (Nc / 64)); };
    const float rsqH = 0.044194173824159216f;   // 1/sqrt(512)

    // 1) x l2-norm -> bf16
    k_normalize<<<dim3(BN / 8), blk, 0, stream>>>(x, xn);
    // 2) weights -> bf16 Bt layouts; E -> bf16; 1/||E_k||
    k_transpose_cvt<<<dim3((Cc * Hh) / 256), blk, 0, stream>>>(w1, w1t, Cc, Hh);
    k_transpose_cvt<<<dim3((Hh * Hh) / 256), blk, 0, stream>>>(w2, w2t, Hh, Hh);
    k_transpose_cvt<<<dim3((Hh * Hh) / 256), blk, 0, stream>>>(wk, wkt, Hh, Hh);
    k_transpose_cvt<<<dim3((Hh * Hh) / 256), blk, 0, stream>>>(wq, wqt, Hh, Hh);
    k_transpose_cvt<<<dim3((Hh * Cc) / 256), blk, 0, stream>>>(wout, woutt, Hh, Cc);
    k_cvt_bf<<<dim3((Kc * Hh) / 256), blk, 0, stream>>>(E, Ebf, Kc * Hh);
    k_enorm<<<dim3(Kc), dim3(32), 0, stream>>>(E, invEn);

    // 3) T1 = relu(xn @ w1 + b1)        [bf16]  (TDM/LDS staged)
    k_gemm_lds<true><<<gbl(BN, Hh), blk, 0, stream>>>(
        xn, Cc, 0, w1t, 0, b1, nullptr, 1.0f,
        nullptr, T1, Hh, 0, BN, Hh, Cc);
    // 4) Hm = T1 @ w2 + b2              [f32 + bf16]  (TDM/LDS staged)
    k_gemm_lds<false><<<gbl(BN, Hh), blk, 0, stream>>>(
        T1, Hh, 0, w2t, 0, b2, nullptr, 1.0f,
        Hm, Hmbf, Hh, 0, BN, Hh, Hh);
    // 5) Kmat = Hm @ wk + wk_b          [bf16]  (TDM/LDS staged; reuses xn)
    k_gemm_lds<false><<<gbl(BN, Hh), blk, 0, stream>>>(
        Hmbf, Hh, 0, wkt, 0, wk_b, nullptr, 1.0f,
        nullptr, Kmatbf, Hh, 0, BN, Hh, Hh);
    // 6) Q = E @ wq + wq_b              [bf16]
    k_gemm<true, false><<<gb(Kc, Hh), blk, 0, stream>>>(
        Ebf, Hh, 0, wqt, Hh, 0, wq_b, nullptr, 1.0f,
        nullptr, Qbf, Hh, 0, nullptr, 0, 0, Kc, Hh, Hh);
    // 7) logits = (Hm @ E^T) * invEn    [f32]  (TDM/LDS staged; panel = all E)
    k_gemm_lds<false><<<gbl(BN, Kc), blk, 0, stream>>>(
        Hmbf, Hh, 0, Ebf, 0, nullptr, invEn, 1.0f,
        Lg, nullptr, Kc, 0, BN, Kc, Hh);
    // 8) P = softmax(logits); M = bern < P
    k_softmax64<<<dim3(BN / 8), blk, 0, stream>>>(Lg, bern, P, Pbf, Mbf);
    // 9) theta = P @ E                  [f32]   (B row-major)
    k_gemm<false, false><<<gb(BN, Hh), blk, 0, stream>>>(
        Pbf, Kc, 0, Ebf, Hh, 0, nullptr, nullptr, 1.0f,
        theta, nullptr, Hh, 0, nullptr, 0, 0, BN, Hh, Kc);
    // 10) scores[b] = Q @ Kmat[b]^T / sqrt(H)   (Bt = Kmat row-major), batched
    k_gemm<true, false><<<dim3(gb(Kc, Nt).x, Bb), blk, 0, stream>>>(
        Qbf, Hh, 0, Kmatbf, Hh, (long)Nt * Hh, nullptr, nullptr, rsqH,
        scores, nullptr, Nt, (long)Kc * Nt, nullptr, 0, 0, Kc, Nt, Hh);
    // 11) A = softmax over N            [bf16]
    k_softmax_row<<<dim3(Bb * Kc), blk, 0, stream>>>(scores, Abf, Nt);
    // 12) C_temp[b] = A[b] @ Hm[b] -> stored transposed bf16 [H][K], batched
    k_gemm<false, false><<<dim3(gb(Kc, Hh).x, Bb), blk, 0, stream>>>(
        Abf, Nt, (long)Kc * Nt, Hmbf, Hh, (long)Nt * Hh, nullptr, nullptr, 1.0f,
        nullptr, nullptr, 0, 0, CtT, Kc, (long)Hh * Kc, Kc, Hh, Nt);
    // 13) C_pre[b] = C_temp[b]^T @ M[b]^T   (TDM/LDS staged, batched)
    k_gemm_lds<false><<<dim3(gbl(Hh, Nt).x, Bb), blk, 0, stream>>>(
        CtT, Kc, (long)Hh * Kc, Mbf, (long)Nt * Kc, nullptr, nullptr, 1.0f,
        Cpre, nullptr, Nt, (long)Hh * Nt, Hh, Nt, Kc);
    // 14) C = l2norm over N, transposed into d_out second half
    k_colnorm<<<dim3(Bb * Hh), blk, 0, stream>>>(Cpre, outC, Nt);
    // 15) U = (Hm + C) * theta          [bf16]  (reuses T1 buffer)
    k_build_U<<<dim3((BN * Hh) / 256), blk, 0, stream>>>(Hm, outC, theta, Ubf);
    // 16) Y = U @ wout + wout_b         [f32 -> d_out first half]  (TDM/LDS)
    k_gemm_lds<false><<<gbl(BN, Cc), blk, 0, stream>>>(
        Ubf, Hh, 0, woutt, 0, wout_b, nullptr, 1.0f,
        Y, nullptr, Cc, 0, BN, Cc, Hh);
}
